// SetToGraphSiam_17351667876251
// MI455X (gfx1250) — matrix operand order repaired and checked by reading, not compiled, not run on hardware
//
#include <hip/hip_runtime.h>
#include <hip/hip_bf16.h>

typedef _Float16 f16;
typedef __attribute__((ext_vector_type(16))) _Float16 v16h;
typedef __attribute__((ext_vector_type(8)))  _Float16 h8;    // 16B chunk
typedef __attribute__((ext_vector_type(8)))  float    v8f;
typedef __attribute__((ext_vector_type(4)))  int      v4i;

#define NPTS 512
#define BATCH 2

// gfx1250 async global->LDS path (ASYNCcnt-tracked); pointer params are
// generic `v4i*` on this toolchain (probed in round 2).
#if defined(__gfx1250__) && __has_builtin(__builtin_amdgcn_global_load_async_to_lds_b128)
#define USE_ASYNC_LDS 1
#else
#define USE_ASYNC_LDS 0
#endif

// ---------------------------------------------------------------------------
// Weight prep: f32 -> f16, padded / split layouts.
//   W0h 128x32 (pad 10->32) | W1h 128x128 | W2h 64x128 | Wah/Wbh 256x64
// ---------------------------------------------------------------------------
__global__ void prep_weights_kernel(const float* __restrict__ Ws0,
                                    const float* __restrict__ Ws1,
                                    const float* __restrict__ Ws2,
                                    const float* __restrict__ Wp0,
                                    f16* __restrict__ W0h, f16* __restrict__ W1h,
                                    f16* __restrict__ W2h, f16* __restrict__ Wah,
                                    f16* __restrict__ Wbh) {
  int idx = blockIdx.x * blockDim.x + threadIdx.x;
  if (idx < 128 * 32) {
    int f = idx >> 5, k = idx & 31;
    W0h[idx] = (k < 10) ? (f16)Ws0[f * 10 + k] : (f16)0.f;
    return;
  }
  idx -= 128 * 32;
  if (idx < 128 * 128) { W1h[idx] = (f16)Ws1[idx]; return; }
  idx -= 128 * 128;
  if (idx < 64 * 128)  { W2h[idx] = (f16)Ws2[idx]; return; }
  idx -= 64 * 128;
  if (idx < 256 * 64) {
    int f = idx >> 6, k = idx & 63;
    Wah[idx] = (f16)Wp0[f * 128 + k];
    return;
  }
  idx -= 256 * 64;
  if (idx < 256 * 64) {
    int f = idx >> 6, k = idx & 63;
    Wbh[idx] = (f16)Wp0[f * 128 + 64 + k];
  }
}

// x (B,N,10) -> U0h (B,32,512) f16, transposed, K padded to 32
__global__ void prep_x_kernel(const float* __restrict__ x, f16* __restrict__ U0h) {
  int idx = blockIdx.x * blockDim.x + threadIdx.x;
  if (idx >= BATCH * 32 * NPTS) return;
  int n = idx & (NPTS - 1);
  int c = (idx >> 9) & 31;
  int b = idx >> 14;
  U0h[idx] = (c < 10) ? (f16)x[(b * NPTS + n) * 10 + c] : (f16)0.f;
}

// ---------------------------------------------------------------------------
// WMMA relu-GEMM layer, LDS-staged:
//   out[b,m,n] = act( sum_k W[m,k] * U[b,k,n] + bias[m] )
// Block = 128 threads (4 waves) -> one 16(M) x 64(N) output tile.
//   * W tile staged into LDS in A-FRAGMENT ORDER (sWf[kstep][lane][16]),
//     so each wave's A fragment is one 32B v16h LDS read (2x ds_load_b128),
//     shared by all 4 waves.
//   * U panel Kx64 staged via global_load_async_to_lds_b128 (ASYNCcnt).
//   * K is a template param -> fully unrolled wmma chain.
// Fragment layouts per CDNA5 ISA 7.12.2:
//   A (16-bit 16x32): m = lane&15; lanes<16 K {0..7,16..23}, lanes>=16 +8.
//   B (32x16):        n = lane&15; same K order as A.
//   C/D:              m = vgpr + 8*(lane>=16), n = lane&15.
// ---------------------------------------------------------------------------
template <int K, bool HAS_BIAS, bool RELU, bool STORE_H, bool STORE_F>
__global__ void wmma_layer_kernel(const f16* __restrict__ W,
                                  const f16* __restrict__ U,
                                  const float* __restrict__ bias,
                                  f16* __restrict__ outH,
                                  float* __restrict__ outF,
                                  int M) {
  __shared__ f16 sWf[16 * K];  // A-fragment-ordered W tile (<= 4 KB)
  __shared__ f16 sU[K * 64];   // row-major U panel (<= 16 KB)

  const int t    = threadIdx.x;       // 0..127
  const int lane = t & 31;
  const int wv   = t >> 5;            // wave id 0..3
  const int hsel = lane >> 4;
  const int l    = lane & 15;
  const int b    = blockIdx.z;
  const int n0   = blockIdx.x << 6;   // 64-wide N tile
  const int tm   = blockIdx.y << 4;
  const int wn   = wv << 4;           // wave's n-subtile inside the 64 tile

  const f16* __restrict__ Wb = W + (size_t)tm * K;
  const f16* __restrict__ Ub = U + (size_t)b * K * NPTS + n0;

  // ---- stage U panel (K rows x 64 halfs) via async global->LDS b128 ----
  #pragma unroll
  for (int idx = t; idx < K * 8; idx += 128) {
    const int k = idx >> 3, c8 = idx & 7;
#if USE_ASYNC_LDS
    __builtin_amdgcn_global_load_async_to_lds_b128(
        (v4i*)(Ub + (size_t)k * NPTS + c8 * 8),
        (v4i*)&sU[k * 64 + c8 * 8], 0, 0);
#else
    *(h8*)&sU[k * 64 + c8 * 8] = *(const h8*)(Ub + (size_t)k * NPTS + c8 * 8);
#endif
  }

  // ---- stage W tile in A-fragment order: sWf[kstep*512 + lane*16 + run*8]
  // Fragment half i of (lane,kstep):  i<8  -> W[row, 32*kstep + 8*hsel + i]
  //                                   i>=8 -> W[row, 32*kstep + 8*hsel + 16 + i-8]
  // Both are 8-half contiguous runs in the source row -> 16B copies.
  #pragma unroll
  for (int c = t; c < 2 * K; c += 128) {   // (K/32 ksteps) * 32 lanes * 2 runs
    const int kstep = c >> 6;
    const int rem   = c & 63;
    const int lf    = rem >> 1;            // target lane 0..31
    const int run   = rem & 1;
    const int row   = lf & 15;
    const int hs    = lf >> 4;
    *(h8*)&sWf[(kstep << 9) + (lf << 4) + (run << 3)] =
        *(const h8*)(Wb + row * K + (kstep << 5) + (hs << 3) + (run << 4));
  }

#if USE_ASYNC_LDS
  asm volatile("s_wait_asynccnt 0" ::: "memory");
#endif
  __syncthreads();

  // ---- K-unrolled WMMA chain ----
  v8f acc = {};
  #pragma unroll
  for (int k0 = 0; k0 < K; k0 += 32) {
    const v16h av = *(const v16h*)&sWf[((k0 >> 5) << 9) + (lane << 4)];
    v16h bv;
    const int kb = k0 + (hsel << 3);
    #pragma unroll
    for (int i = 0; i < 16; ++i)
      bv[i] = sU[(kb + (i & 7) + ((i >> 3) << 4)) * 64 + wn + l];

    acc = __builtin_amdgcn_wmma_f32_16x16x32_f16(
        false, av, false, bv, (short)0, acc, false, false);
  }

  // ---- branch-free epilogue ----
  #pragma unroll
  for (int v = 0; v < 8; ++v) {
    const int m = tm + v + (hsel << 3);
    float r = acc[v];
    if constexpr (HAS_BIAS) r += bias[m];
    if constexpr (RELU)     r = fmaxf(r, 0.f);
    const size_t o = (size_t)b * M * NPTS + (size_t)m * NPTS + n0 + wn + l;
    if constexpr (STORE_H)  outH[o] = (f16)r;
    if constexpr (STORE_F)  outF[o] = r;
  }
}

// ---------------------------------------------------------------------------
// Fused bilinear-relu reduction (h never materialized):
//   out[b,i,j] = sum_c w[c]*relu(ha[b,c,j] + hb[b,c,i]) + bp1
// (bp0 folded into ha during its GEMM.) 32x32 (i,j) tile, 64-channel LDS
// chunks; sha reads lane-contiguous, shb/sw broadcast.
// ---------------------------------------------------------------------------
__global__ void pairwise_out_kernel(const float* __restrict__ ha,
                                    const float* __restrict__ hb,
                                    const float* __restrict__ w,
                                    const float* __restrict__ bp1,
                                    float* __restrict__ out) {
  __shared__ float sha[64][32];
  __shared__ float shb[64][32];
  __shared__ float sw[64];

  const int t  = threadIdx.x;          // 256 threads
  const int b  = blockIdx.z;
  const int i0 = blockIdx.y << 5;
  const int j0 = blockIdx.x << 5;

  const float* __restrict__ hab = ha + (size_t)b * 256 * NPTS;
  const float* __restrict__ hbb = hb + (size_t)b * 256 * NPTS;

  const int j  = t & 31;
  const int ib = t >> 5;               // 0..7

  float acc0 = 0.f, acc1 = 0.f, acc2 = 0.f, acc3 = 0.f;

  for (int c0 = 0; c0 < 256; c0 += 64) {
    __syncthreads();
    #pragma unroll
    for (int idx = t; idx < 64 * 32; idx += 256) {
      const int c = idx >> 5, q = idx & 31;
      sha[c][q] = hab[(size_t)(c0 + c) * NPTS + j0 + q];
      shb[c][q] = hbb[(size_t)(c0 + c) * NPTS + i0 + q];
    }
    if (t < 64) sw[t] = w[c0 + t];
    __syncthreads();

    #pragma unroll 4
    for (int c = 0; c < 64; ++c) {
      const float av = sha[c][j];
      const float wc = sw[c];
      acc0 += wc * fmaxf(av + shb[c][ib],      0.f);
      acc1 += wc * fmaxf(av + shb[c][ib + 8],  0.f);
      acc2 += wc * fmaxf(av + shb[c][ib + 16], 0.f);
      acc3 += wc * fmaxf(av + shb[c][ib + 24], 0.f);
    }
  }

  const float bb = *bp1;
  const size_t base = (size_t)b * NPTS * NPTS + (size_t)j0 + j;
  out[base + (size_t)(i0 + ib)      * NPTS] = acc0 + bb;
  out[base + (size_t)(i0 + ib + 8)  * NPTS] = acc1 + bb;
  out[base + (size_t)(i0 + ib + 16) * NPTS] = acc2 + bb;
  out[base + (size_t)(i0 + ib + 24) * NPTS] = acc3 + bb;
}

// ---------------------------------------------------------------------------
extern "C" void kernel_launch(void* const* d_in, const int* in_sizes, int n_in,
                              void* d_out, int out_size, void* d_ws, size_t ws_size,
                              hipStream_t stream) {
  const float* x   = (const float*)d_in[0];
  const float* Ws0 = (const float*)d_in[1];
  const float* bs0 = (const float*)d_in[2];
  const float* Ws1 = (const float*)d_in[3];
  const float* bs1 = (const float*)d_in[4];
  const float* Ws2 = (const float*)d_in[5];
  const float* bs2 = (const float*)d_in[6];
  const float* Wp0 = (const float*)d_in[7];
  const float* bp0 = (const float*)d_in[8];
  const float* Wp1 = (const float*)d_in[9];
  const float* bp1 = (const float*)d_in[10];
  float* out = (float*)d_out;

  char* ws = (char*)d_ws;
  f16*   W0h = (f16*)(ws + 0);          //  128*32*2   =   8192
  f16*   W1h = (f16*)(ws + 8192);       //  128*128*2  =  32768
  f16*   W2h = (f16*)(ws + 40960);      //   64*128*2  =  16384
  f16*   Wah = (f16*)(ws + 57344);      //  256*64*2   =  32768
  f16*   Wbh = (f16*)(ws + 90112);      //  256*64*2   =  32768
  f16*   U0h = (f16*)(ws + 122880);     // 2*32*512*2  =  65536
  f16*   U1h = (f16*)(ws + 188416);     // 2*128*512*2 = 262144
  f16*   U2h = (f16*)(ws + 450560);     // 2*128*512*2 = 262144
  f16*   U3h = (f16*)(ws + 712704);     // 2*64*512*2  = 131072
  float* HAf = (float*)(ws + 843776);   // 2*256*512*4 = 1048576
  float* HBf = (float*)(ws + 1892352);  // 2*256*512*4 = 1048576

  prep_weights_kernel<<<240, 256, 0, stream>>>(Ws0, Ws1, Ws2, Wp0,
                                               W0h, W1h, W2h, Wah, Wbh);
  prep_x_kernel<<<128, 256, 0, stream>>>(x, U0h);

  // Set encoder: three relu-GEMMs on the WMMA path
  wmma_layer_kernel< 32, true, true, true, false>
      <<<dim3(NPTS / 64,  8, BATCH), 128, 0, stream>>>(W0h, U0h, bs0, U1h, nullptr, 128);
  wmma_layer_kernel<128, true, true, true, false>
      <<<dim3(NPTS / 64,  8, BATCH), 128, 0, stream>>>(W1h, U1h, bs1, U2h, nullptr, 128);
  wmma_layer_kernel<128, true, true, true, false>
      <<<dim3(NPTS / 64,  4, BATCH), 128, 0, stream>>>(W2h, U2h, bs2, U3h, nullptr, 64);

  // ha = Wa u + bp0 (no relu), hb = Wb u  -> f32 workspace
  wmma_layer_kernel< 64, true, false, false, true>
      <<<dim3(NPTS / 64, 16, BATCH), 128, 0, stream>>>(Wah, U3h, bp0, nullptr, HAf, 256);
  wmma_layer_kernel< 64, false, false, false, true>
      <<<dim3(NPTS / 64, 16, BATCH), 128, 0, stream>>>(Wbh, U3h, nullptr, nullptr, HBf, 256);

  // Fused bilinear relu reduction -> out (B,1,512,512)
  pairwise_out_kernel<<<dim3(NPTS / 32, NPTS / 32, BATCH), 256, 0, stream>>>(
      HAf, HBf, Wp1, bp1, out);
}
